// Encoder_Decoder_23416161697847
// MI455X (gfx1250) — compile-verified, matrix operand order checked by reading, not run on hardware
//
#include <hip/hip_runtime.h>
#include <hip/hip_bf16.h>

// MI455X / gfx1250, wave32. All big GEMMs via v_wmma_f32_16x16x32_bf16.

typedef __bf16 bf16_t;
typedef __attribute__((ext_vector_type(16))) __bf16 v16bf;
typedef __attribute__((ext_vector_type(8)))  float  v8f;

#define NN   8192
#define SS   4
#define ZZ   64
#define HH   128
#define KK   4
#define DEGN 16
#define RR   (NN*SS)          // 32768 rows (n,s)

// ---------------- WMMA fragment helpers (ISA 7.12.2 layouts) ----------------
// A / Bt fragment: 16 rows x 32 k, row-major source, leading dim `ld`.
// lane L: row m = L&15 ; k-half hb = (L>>4)*8 ; elem i<8 -> k=hb+i ; i>=8 -> k=16+hb+(i-8)
__device__ __forceinline__ v16bf frag_bf16(const bf16_t* base, int ld) {
  const int lane = threadIdx.x & 31;
  const bf16_t* r = base + (lane & 15) * ld + ((lane >> 4) << 3);
  v16bf f;
#pragma unroll
  for (int i = 0; i < 8; ++i) { f[i] = r[i]; f[i + 8] = r[i + 16]; }
  return f;
}
__device__ __forceinline__ v16bf frag_f32(const float* base, int ld) {
  const int lane = threadIdx.x & 31;
  const float* r = base + (lane & 15) * ld + ((lane >> 4) << 3);
  v16bf f;
#pragma unroll
  for (int i = 0; i < 8; ++i) { f[i] = (__bf16)r[i]; f[i + 8] = (__bf16)r[i + 16]; }
  return f;
}
__device__ __forceinline__ v8f wmma_bf(v16bf a, v16bf b, v8f c) {
  return __builtin_amdgcn_wmma_f32_16x16x32_bf16(false, a, false, b, (short)0, c, false, false);
}
// C/D f32: lane L col n=L&15, VGPR v -> row m = (L>>4)*8 + v
__device__ __forceinline__ void store_frag_bf16(bf16_t* base, int ld, v8f c) {
  const int lane = threadIdx.x & 31;
  const int n = lane & 15, mb = (lane >> 4) << 3;
#pragma unroll
  for (int v = 0; v < 8; ++v) base[(size_t)(mb + v) * ld + n] = (__bf16)c[v];
}
__device__ __forceinline__ void store_frag_f32(float* base, int ld, v8f c) {
  const int lane = threadIdx.x & 31;
  const int n = lane & 15, mb = (lane >> 4) << 3;
#pragma unroll
  for (int v = 0; v < 8; ++v) base[(size_t)(mb + v) * ld + n] = c[v];
}

// ---------------- K0: weight prep (bf16 copies) + G_k = Ws_k^T Wd_k --------
__global__ __launch_bounds__(256) void k0_prep(
    const float* __restrict__ Ws, const float* __restrict__ Wd,
    const float* __restrict__ F2w1, const float* __restrict__ F2w2,
    const float* __restrict__ F1w1, const float* __restrict__ F1w2,
    bf16_t* __restrict__ G, bf16_t* __restrict__ W1a, bf16_t* __restrict__ W1b,
    bf16_t* __restrict__ W2, bf16_t* __restrict__ F1w1b, bf16_t* __restrict__ F1w2b) {
  const int tid = blockIdx.x * blockDim.x + threadIdx.x;   // 32768 threads
  if (tid < 16384) {
    const int o = tid >> 7, i = tid & 127;
    W1a[tid]   = (__bf16)F2w1[o * 256 + i];        // F2_w1[:, :128]
    W1b[tid]   = (__bf16)F2w1[o * 256 + 128 + i];  // F2_w1[:, 128:]
    W2[tid]    = (__bf16)F2w2[tid];
    F1w2b[tid] = (__bf16)F1w2[tid];
    // G[k][z1][z2] = sum_h Ws[k,h,z1]*Wd[k,h,z2]
    const int k = tid >> 12, z1 = (tid >> 6) & 63, z2 = tid & 63;
    const float* a = Ws + (size_t)k * HH * ZZ + z1;
    const float* b = Wd + (size_t)k * HH * ZZ + z2;
    float acc = 0.f;
#pragma unroll 8
    for (int h = 0; h < HH; ++h) acc += a[h * ZZ] * b[h * ZZ];
    G[tid] = (__bf16)acc;
  }
  for (int i = tid; i < 128 * 512; i += 32768) F1w1b[i] = (__bf16)F1w1[i];
}

// ---------------- K1: per-row-tile node GEMMs: P, Q, V ----------------------
// P[r,o]=x[r]@W1a_row_o ; Q[r,o]=x[r]@W1b_row_o ; V[r,k*64+z1]=sum_z2 G[k][z1][z2]*z[r,z2]
__global__ __launch_bounds__(128) void k1_node(
    const float* __restrict__ zIG, const float* __restrict__ xt,
    const bf16_t* __restrict__ G, const bf16_t* __restrict__ W1a,
    const bf16_t* __restrict__ W1b,
    bf16_t* __restrict__ P, bf16_t* __restrict__ Q, bf16_t* __restrict__ V) {
  const int r0 = blockIdx.x * 16;
  const int w  = threadIdx.x >> 5;          // wave 0..3
  {
    const int c0 = 2 * w, c1 = 2 * w + 1;   // coltiles of 8 (H=128)
    v8f p0 = {}, p1 = {}, q0 = {}, q1 = {};
#pragma unroll
    for (int kc = 0; kc < 4; ++kc) {
      v16bf a  = frag_f32(xt + (size_t)r0 * HH + kc * 32, HH);
      v16bf b0 = frag_bf16(W1a + (size_t)c0 * 16 * HH + kc * 32, HH);
      v16bf b1 = frag_bf16(W1a + (size_t)c1 * 16 * HH + kc * 32, HH);
      p0 = wmma_bf(a, b0, p0);  p1 = wmma_bf(a, b1, p1);
      v16bf b2 = frag_bf16(W1b + (size_t)c0 * 16 * HH + kc * 32, HH);
      v16bf b3 = frag_bf16(W1b + (size_t)c1 * 16 * HH + kc * 32, HH);
      q0 = wmma_bf(a, b2, q0);  q1 = wmma_bf(a, b3, q1);
    }
    store_frag_bf16(P + (size_t)r0 * HH + c0 * 16, HH, p0);
    store_frag_bf16(P + (size_t)r0 * HH + c1 * 16, HH, p1);
    store_frag_bf16(Q + (size_t)r0 * HH + c0 * 16, HH, q0);
    store_frag_bf16(Q + (size_t)r0 * HH + c1 * 16, HH, q1);
  }
#pragma unroll
  for (int pr = 0; pr < 2; ++pr) {          // V: 16 coltiles of 256 cols, 4/wave
    const int c0 = 4 * w + 2 * pr, c1 = c0 + 1;
    const int k0 = (c0 * 16) >> 6, z0 = (c0 * 16) & 63;
    const int k1 = (c1 * 16) >> 6, z1 = (c1 * 16) & 63;
    v8f v0 = {}, v1 = {};
#pragma unroll
    for (int kc = 0; kc < 2; ++kc) {
      v16bf a  = frag_f32(zIG + (size_t)r0 * ZZ + kc * 32, ZZ);
      v16bf b0 = frag_bf16(G + (size_t)k0 * 4096 + z0 * ZZ + kc * 32, ZZ);
      v16bf b1 = frag_bf16(G + (size_t)k1 * 4096 + z1 * ZZ + kc * 32, ZZ);
      v0 = wmma_bf(a, b0, v0);  v1 = wmma_bf(a, b1, v1);
    }
    store_frag_bf16(V + (size_t)r0 * 256 + c0 * 16, 256, v0);
    store_frag_bf16(V + (size_t)r0 * 256 + c1 * 16, 256, v1);
  }
}

// ---------------- K2: fused per-node edge stage -----------------------------
// scores -> leaky_relu -> softmax(d) [alpha out] ; hidden=relu(p[src]+q[n]) ;
// dA = hidden @ W2^T (WMMA) ; res[s,k,o] = mean_d alpha*dA -> RES (bf16)
__global__ __launch_bounds__(128) void k2_edge(
    const float* __restrict__ zIG, const int* __restrict__ srcIdx,
    const bf16_t* __restrict__ P, const bf16_t* __restrict__ Q,
    const bf16_t* __restrict__ V, const bf16_t* __restrict__ W2,
    bf16_t* __restrict__ RES, float* __restrict__ alphaOut) {
  __shared__ int    ssrc[16];
  __shared__ float  qn[4][128];
  __shared__ float  sc[256];            // [d*16 + s*4 + k]
  __shared__ float  al[256];
  __shared__ bf16_t Hid[64][128];       // row = d*4+s
  __shared__ float  dAls[64][128];

  const int n = blockIdx.x, t = threadIdx.x;
  if (t < 16) ssrc[t] = srcIdx[n * DEGN + t];
  for (int i = t; i < 512; i += 128)
    qn[i >> 7][i & 127] = (float)Q[((size_t)n * 4 + (i >> 7)) * HH + (i & 127)];
  __syncthreads();

  for (int id = t; id < 256; id += 128) {           // edge scores (VALU dots)
    const int d = id >> 4, s = (id >> 2) & 3, k = id & 3;
    const float*  zp = zIG + ((size_t)ssrc[d] * 4 + s) * ZZ;
    const bf16_t* vp = V + ((size_t)n * 4 + s) * 256 + k * 64;
    float acc = 0.f;
#pragma unroll
    for (int z = 0; z < 64; ++z) acc += zp[z] * (float)vp[z];
    sc[d * 16 + s * 4 + k] = (acc > 0.f) ? acc : 0.01f * acc;  // leaky_relu
  }
  __syncthreads();

  if (t < 16) {                                     // softmax over d per (s,k)
    const int s = t >> 2, k = t & 3;
    float mx = -1e30f;
#pragma unroll
    for (int d = 0; d < 16; ++d) mx = fmaxf(mx, sc[d * 16 + s * 4 + k]);
    float e[16], sum = 0.f;
#pragma unroll
    for (int d = 0; d < 16; ++d) { e[d] = __expf(sc[d * 16 + s * 4 + k] - mx); sum += e[d]; }
    const float inv = 1.f / sum;
#pragma unroll
    for (int d = 0; d < 16; ++d) {
      const float a = e[d] * inv;
      al[d * 16 + s * 4 + k] = a;
      alphaOut[(((size_t)n * 16 + d) * 4 + s) * 4 + k] = a;   // [N,D,S,K]
    }
  }
  for (int i = t; i < 64 * 128; i += 128) {         // hidden = relu(p_src + q_dst)
    const int row = i >> 7, col = i & 127, d = row >> 2, s = row & 3;
    const float hv = (float)P[((size_t)ssrc[d] * 4 + s) * HH + col] + qn[s][col];
    Hid[row][col] = (__bf16)(hv > 0.f ? hv : 0.f);
  }
  __syncthreads();

  const int w = t >> 5;                             // dA GEMM: wave w -> rows 16w..16w+15
#pragma unroll
  for (int pc = 0; pc < 4; ++pc) {
    const int c0 = 2 * pc, c1 = 2 * pc + 1;
    v8f a0 = {}, a1 = {};
#pragma unroll
    for (int kc = 0; kc < 4; ++kc) {
      v16bf a  = frag_bf16(&Hid[w * 16][kc * 32], 128);
      v16bf b0 = frag_bf16(W2 + (size_t)c0 * 16 * HH + kc * 32, HH);
      v16bf b1 = frag_bf16(W2 + (size_t)c1 * 16 * HH + kc * 32, HH);
      a0 = wmma_bf(a, b0, a0);  a1 = wmma_bf(a, b1, a1);
    }
    const int lane = t & 31, nn_ = lane & 15, mb = (lane >> 4) << 3;
#pragma unroll
    for (int v = 0; v < 8; ++v) {
      dAls[w * 16 + mb + v][c0 * 16 + nn_] = a0[v];
      dAls[w * 16 + mb + v][c1 * 16 + nn_] = a1[v];
    }
  }
  __syncthreads();

  for (int id = t; id < 2048; id += 128) {          // weighted mean over d
    const int s = id >> 9, k = (id >> 7) & 3, o = id & 127;
    float acc = 0.f;
#pragma unroll
    for (int d = 0; d < 16; ++d) acc += al[d * 16 + s * 4 + k] * dAls[d * 4 + s][o];
    RES[((size_t)n * 4 + s) * 512 + k * 128 + o] = (__bf16)(acc * (1.f / 16.f));
  }
}

// ---------------- K3: deltax = relu(res @ F1_w1^T) @ F1_w2^T ----------------
__global__ __launch_bounds__(128) void k3_final(
    const bf16_t* __restrict__ RES, const bf16_t* __restrict__ F1w1b,
    const bf16_t* __restrict__ F1w2b, float* __restrict__ deltax) {
  __shared__ bf16_t T[16][128];
  const int r0 = blockIdx.x * 16;
  const int w  = threadIdx.x >> 5;
  const int c0 = 2 * w, c1 = 2 * w + 1;
  v8f a0 = {}, a1 = {};
#pragma unroll
  for (int kc = 0; kc < 16; ++kc) {                 // K = 512
    v16bf a  = frag_bf16(RES + (size_t)r0 * 512 + kc * 32, 512);
    v16bf b0 = frag_bf16(F1w1b + (size_t)c0 * 16 * 512 + kc * 32, 512);
    v16bf b1 = frag_bf16(F1w1b + (size_t)c1 * 16 * 512 + kc * 32, 512);
    a0 = wmma_bf(a, b0, a0);  a1 = wmma_bf(a, b1, a1);
  }
  {
    const int lane = threadIdx.x & 31, nn_ = lane & 15, mb = (lane >> 4) << 3;
#pragma unroll
    for (int v = 0; v < 8; ++v) {                   // relu -> bf16 staging
      const float x0 = a0[v], x1 = a1[v];
      T[mb + v][c0 * 16 + nn_] = (__bf16)(x0 > 0.f ? x0 : 0.f);
      T[mb + v][c1 * 16 + nn_] = (__bf16)(x1 > 0.f ? x1 : 0.f);
    }
  }
  __syncthreads();
  v8f d0 = {}, d1 = {};
#pragma unroll
  for (int kc = 0; kc < 4; ++kc) {                  // K = 128
    v16bf a  = frag_bf16(&T[0][kc * 32], 128);
    v16bf b0 = frag_bf16(F1w2b + (size_t)c0 * 16 * HH + kc * 32, HH);
    v16bf b1 = frag_bf16(F1w2b + (size_t)c1 * 16 * HH + kc * 32, HH);
    d0 = wmma_bf(a, b0, d0);  d1 = wmma_bf(a, b1, d1);
  }
  store_frag_f32(deltax + (size_t)r0 * HH + c0 * 16, HH, d0);
  store_frag_f32(deltax + (size_t)r0 * HH + c1 * 16, HH, d1);
}

// ---------------- launch ----------------------------------------------------
extern "C" void kernel_launch(void* const* d_in, const int* in_sizes, int n_in,
                              void* d_out, int out_size, void* d_ws, size_t ws_size,
                              hipStream_t stream) {
  const float* zIG  = (const float*)d_in[0];
  const float* xt   = (const float*)d_in[1];
  const float* Ws   = (const float*)d_in[2];
  const float* Wd   = (const float*)d_in[3];
  const float* F2w1 = (const float*)d_in[4];
  const float* F2w2 = (const float*)d_in[5];
  const float* F1w1 = (const float*)d_in[6];
  const float* F1w2 = (const float*)d_in[7];
  const int*   src  = (const int*)d_in[8];
  (void)in_sizes; (void)n_in; (void)out_size; (void)ws_size;

  float* deltax   = (float*)d_out;                       // [N,S,H]
  float* alphaOut = (float*)d_out + (size_t)RR * HH;     // [N,D,S,K]

  // workspace layout (bf16)
  char* w = (char*)d_ws;
  size_t off = 0;
  bf16_t* G     = (bf16_t*)(w + off); off += (size_t)KK * 64 * 64 * 2;
  bf16_t* W1a   = (bf16_t*)(w + off); off += (size_t)128 * 128 * 2;
  bf16_t* W1b   = (bf16_t*)(w + off); off += (size_t)128 * 128 * 2;
  bf16_t* W2    = (bf16_t*)(w + off); off += (size_t)128 * 128 * 2;
  bf16_t* F1w1b = (bf16_t*)(w + off); off += (size_t)128 * 512 * 2;
  bf16_t* F1w2b = (bf16_t*)(w + off); off += (size_t)128 * 128 * 2;
  bf16_t* P     = (bf16_t*)(w + off); off += (size_t)RR * 128 * 2;
  bf16_t* Q     = (bf16_t*)(w + off); off += (size_t)RR * 128 * 2;
  bf16_t* V     = (bf16_t*)(w + off); off += (size_t)RR * 256 * 2;
  bf16_t* RES   = (bf16_t*)(w + off); off += (size_t)RR * 512 * 2;   // ~67 MB total

  k0_prep<<<128, 256, 0, stream>>>(Ws, Wd, F2w1, F2w2, F1w1, F1w2,
                                   G, W1a, W1b, W2, F1w1b, F1w2b);
  k1_node<<<RR / 16, 128, 0, stream>>>(zIG, xt, G, W1a, W1b, P, Q, V);
  k2_edge<<<NN, 128, 0, stream>>>(zIG, src, P, Q, V, W2, RES, alphaOut);
  k3_final<<<RR / 16, 128, 0, stream>>>(RES, F1w1b, F1w2b, deltax);
}
